// Llama4TextMoe_37589553774973
// MI455X (gfx1250) — matrix operand order repaired and checked by reading, not compiled, not run on hardware
//
#include <hip/hip_runtime.h>

// ---------------------------------------------------------------------------
// Llama4-style MoE (top-1) + shared expert, bf16 WMMA pipeline for gfx1250.
// H=2048, I=4096, E=8, T=2048.  out = [T*H] ++ router_scores [E*T], f32.
//
// Roofline: ~900 MB of f32 weights/activations (each expert's weights read
// exactly once -> tokens grouped per expert) ~= 38.6 us at 23.3 TB/s;
// ~206 GFLOP at bf16 WMMA rate ~= 41 us. Balanced => stream f32 weights with
// NT hint, convert to bf16 in flight, f32 WMMA accumulation, double-buffered
// LDS so global->LDS staging overlaps v_wmma.
// ---------------------------------------------------------------------------

typedef unsigned short ushort_t;
typedef unsigned int   uint32;

typedef __attribute__((ext_vector_type(16))) __bf16 v16bf;
typedef __attribute__((ext_vector_type(8)))  float  v8f;
typedef __attribute__((ext_vector_type(4)))  float  f4;

static constexpr int Hdim = 2048;
static constexpr int Idim = 4096;
static constexpr int Tn   = 2048;
static constexpr int Ne   = 8;

union Frag16 { v16bf v; uint4 q[2]; };

__device__ __forceinline__ ushort_t f2bf(float f) {
  // round-to-nearest-even f32 -> bf16
  uint32 u = __float_as_uint(f);
  u += 0x7FFFu + ((u >> 16) & 1u);
  return (ushort_t)(u >> 16);
}

// ---------------------------------------------------------------------------
// 1) Router: logits = x @ router_w, top-1, sigmoid; write [E,T] scores
//    (zeros elsewhere == sigmoid(-inf)), plus per-token expert id / score.
// ---------------------------------------------------------------------------
__global__ __launch_bounds__(256) void router_kernel(
    const float* __restrict__ hs, const float* __restrict__ rw,
    float* __restrict__ scores, int* __restrict__ eid, float* __restrict__ sval)
{
  int wave = threadIdx.x >> 5;
  int lane = threadIdx.x & 31;
  int t = blockIdx.x * 8 + wave;
  const float* x = hs + (size_t)t * Hdim;

  float acc[Ne];
#pragma unroll
  for (int e = 0; e < Ne; ++e) acc[e] = 0.f;
  for (int h = lane; h < Hdim; h += 32) {
    float xv = x[h];
    const float* w = rw + (size_t)h * Ne;
#pragma unroll
    for (int e = 0; e < Ne; ++e) acc[e] += xv * w[e];
  }
#pragma unroll
  for (int e = 0; e < Ne; ++e) {
    for (int off = 16; off; off >>= 1)
      acc[e] += __shfl_xor(acc[e], off, 32);
  }
  int bi = 0; float bv = acc[0];
#pragma unroll
  for (int e = 1; e < Ne; ++e) { if (acc[e] > bv) { bv = acc[e]; bi = e; } }
  float sig = 1.f / (1.f + __expf(-bv));
  if (lane < Ne) scores[(size_t)lane * Tn + t] = (lane == bi) ? sig : 0.f;
  if (lane == 0) { eid[t] = bi; sval[t] = sig; }
}

// ---------------------------------------------------------------------------
// 2) Deterministic stable compaction: wave w builds token list of expert w
//    with ballot/prefix-popcount (order == token order -> deterministic).
// ---------------------------------------------------------------------------
__global__ __launch_bounds__(256) void compact_kernel(
    const int* __restrict__ eid, int* __restrict__ lists, int* __restrict__ counts)
{
  int wave = threadIdx.x >> 5;     // expert handled by this wave
  int lane = threadIdx.x & 31;
  int offset = 0;
  for (int base = 0; base < Tn; base += 32) {
    int t = base + lane;
    bool m = (eid[t] == wave);
    uint32 mask = (uint32)__ballot(m);
    if (m) {
      int pos = __popc(mask & ((1u << lane) - 1u));
      lists[wave * Tn + offset + pos] = t;
    }
    offset += __popc(mask);
  }
  if (lane == 0) counts[wave] = offset;
}

// ---------------------------------------------------------------------------
// 3) f32 -> bf16 conversion of activations (vectorized: b128 in, b64 out).
// ---------------------------------------------------------------------------
__global__ __launch_bounds__(256) void cvt_kernel(
    const float* __restrict__ in, ushort_t* __restrict__ out, int n4)
{
  int i = blockIdx.x * blockDim.x + threadIdx.x;
  int stride = gridDim.x * blockDim.x;
  for (; i < n4; i += stride) {
    f4 v = *(const f4*)(in + (size_t)i * 4);
    uint2 p;
    p.x = (uint32)f2bf(v.x) | ((uint32)f2bf(v.y) << 16);
    p.y = (uint32)f2bf(v.z) | ((uint32)f2bf(v.w) << 16);
    *(uint2*)(out + (size_t)i * 4) = p;
  }
}

// ---------------------------------------------------------------------------
// 4) Fused gate/up GEMM + SwiGLU epilogue.
//    Block tile 128(M)x128(N), K-step 32, 8 waves (4x2), wave tile 32x64.
//    Double-buffered LDS; weights loaded as NT float4, converted to bf16,
//    stored K-transposed so fragments are 2x ds_load_b128.
//    gridDim.y = expert (weights/lists offset in-kernel; 1 for shared).
// ---------------------------------------------------------------------------
__global__ __launch_bounds__(256) void gateup_kernel(
    const ushort_t* __restrict__ Xb,
    const float* __restrict__ Wg0, const float* __restrict__ Wu0,
    size_t expertStride, int ldw,
    ushort_t* __restrict__ Hout,
    const int* __restrict__ listBase, const float* __restrict__ sArr,
    const int* __restrict__ countBase)
{
  __shared__ alignas(16) ushort_t lA [2 * 128 * 32];
  __shared__ alignas(16) ushort_t lBg[2 * 128 * 32];
  __shared__ alignas(16) ushort_t lBu[2 * 128 * 32];
  __shared__ int   rIdx[128];
  __shared__ float rS  [128];

  const int e = blockIdx.y;
  const int* rowList = listBase ? (listBase + (size_t)e * Tn) : nullptr;
  const int count = countBase ? countBase[e] : Tn;
  const float* Wg = Wg0 + (size_t)e * expertStride;
  const float* Wu = Wu0 + (size_t)e * expertStride;

  const int Ntiles = Idim / 128;                    // 32
  int mt = (int)blockIdx.x / Ntiles;
  int nt = (int)blockIdx.x % Ntiles;
  int m0 = mt * 128;
  if (m0 >= count) return;                          // uniform early exit
  int n0 = nt * 128;

  int tid = threadIdx.x;
  if (tid < 128) {
    int gr = m0 + tid;
    int row; float s;
    if (gr < count) {
      row = rowList ? rowList[gr] : gr;
      s   = sArr ? sArr[row] : 1.0f;
    } else {                                        // pad rows: valid addr, zero scale
      row = rowList ? rowList[0] : 0;
      s   = 0.0f;
    }
    rIdx[tid] = row; rS[tid] = s;
  }
  __syncthreads();

  auto load_tiles = [&](int kt, int buf) {
    const int k0 = kt * 32;
    ushort_t* A  = lA  + buf * (128 * 32);
    ushort_t* Bg = lBg + buf * (128 * 32);
    ushort_t* Bu = lBu + buf * (128 * 32);
    // A: 128x32 bf16 (row-gathered), 512 uint4s -> 2 per thread
#pragma unroll
    for (int it = 0; it < 2; ++it) {
      int idx = tid + it * 256;
      int m = idx >> 2, q = (idx & 3) * 8;
      *(uint4*)(A + m * 32 + q) =
          *(const uint4*)(Xb + (size_t)rIdx[m] * Hdim + (size_t)(k0 + q));
    }
    // B: 32x128 f32 streamed NT as float4, bf16-converted, transposed to [n][k]
#pragma unroll
    for (int it = 0; it < 4; ++it) {
      int idx = tid + it * 256;
      int k = idx >> 5, n4 = (idx & 31) * 4;
      size_t go = (size_t)(k0 + k) * (size_t)ldw + (size_t)(n0 + n4);
      f4 vg = __builtin_nontemporal_load((const f4*)(Wg + go));
      f4 vu = __builtin_nontemporal_load((const f4*)(Wu + go));
      ushort_t* bg = Bg + n4 * 32 + k;
      ushort_t* bu = Bu + n4 * 32 + k;
      bg[0] = f2bf(vg.x); bg[32] = f2bf(vg.y); bg[64] = f2bf(vg.z); bg[96] = f2bf(vg.w);
      bu[0] = f2bf(vu.x); bu[32] = f2bf(vu.y); bu[64] = f2bf(vu.z); bu[96] = f2bf(vu.w);
    }
  };

  int wave = tid >> 5, lane = tid & 31;
  int wm = wave & 3, wn = wave >> 2;                // 4 x 2 wave grid
  int g = lane >> 4, ln = lane & 15;

  const v8f vzero = {};
  v8f G[2][4], U[2][4];
#pragma unroll
  for (int i = 0; i < 2; ++i)
#pragma unroll
    for (int j = 0; j < 4; ++j) { G[i][j] = vzero; U[i][j] = vzero; }

  const int KT = Hdim / 32;                         // 64 K-steps
  load_tiles(0, 0);
  __syncthreads();
  for (int kt = 0; kt < KT; ++kt) {
    int cur = kt & 1;
    if (kt + 1 < KT) load_tiles(kt + 1, cur ^ 1);   // overlap with WMMA below

    const ushort_t* A  = lA  + cur * (128 * 32);
    const ushort_t* Bg = lBg + cur * (128 * 32);
    const ushort_t* Bu = lBu + cur * (128 * 32);

    Frag16 bg[4], bu[4];
#pragma unroll
    for (int j = 0; j < 4; ++j) {
      int nc = wn * 64 + j * 16 + ln;
      const uint4* pg = (const uint4*)(Bg + nc * 32);
      const uint4* pu = (const uint4*)(Bu + nc * 32);
      bg[j].q[0] = pg[2 * g]; bg[j].q[1] = pg[2 * g + 1];   // K = 16g..16g+15
      bu[j].q[0] = pu[2 * g]; bu[j].q[1] = pu[2 * g + 1];
    }
#pragma unroll
    for (int i2 = 0; i2 < 2; ++i2) {
      Frag16 a;
      int mr = wm * 32 + i2 * 16 + ln;
      const uint4* pa = (const uint4*)(A + mr * 32);
      a.q[0] = pa[2 * g];        // K = 16g..16g+7
      a.q[1] = pa[2 * g + 1];    // K = 16g+8..16g+15
#pragma unroll
      for (int j = 0; j < 4; ++j) {
        G[i2][j] = __builtin_amdgcn_wmma_f32_16x16x32_bf16(
            false, a.v, false, bg[j].v, (short)0, G[i2][j], false, false);
        U[i2][j] = __builtin_amdgcn_wmma_f32_16x16x32_bf16(
            false, a.v, false, bu[j].v, (short)0, U[i2][j], false, false);
      }
    }
    __syncthreads();
  }

  // SwiGLU epilogue: score applied pre-silu (== scaling the GEMM input)
#pragma unroll
  for (int i2 = 0; i2 < 2; ++i2) {
    int mb = wm * 32 + i2 * 16 + g * 8;
#pragma unroll
    for (int r = 0; r < 8; ++r) {
      int m = mb + r;
      if (m0 + m < count) {
        int row = rIdx[m]; float s = rS[m];
        size_t base = (size_t)row * Idim + (size_t)(n0 + wn * 64 + ln);
#pragma unroll
        for (int j = 0; j < 4; ++j) {
          float gg = s * G[i2][j][r];
          float uu = s * U[i2][j][r];
          float hv = gg * uu / (1.f + __expf(-gg));   // silu(g)*u
          Hout[base + (size_t)(j * 16)] = f2bf(hv);
        }
      }
    }
  }
}

// ---------------------------------------------------------------------------
// 5) Down-projection GEMM: Out[row, :H] (+)= Hm[row, :I] @ Wd[I, H].
//    Same tiling/double-buffering. accumulate=0 writes (shared pass covers
//    all rows first), accumulate=1 adds (each token has exactly one expert
//    -> no conflicts; kernels serialize on the stream).
// ---------------------------------------------------------------------------
__global__ __launch_bounds__(256) void down_kernel(
    const ushort_t* __restrict__ Hm,
    const float* __restrict__ Wd0, size_t expertStride,
    float* __restrict__ Out,
    const int* __restrict__ listBase, const int* __restrict__ countBase,
    int accumulate)
{
  __shared__ alignas(16) ushort_t lA[2 * 128 * 32];
  __shared__ alignas(16) ushort_t lB[2 * 128 * 32];
  __shared__ int rIdx[128];

  const int e = blockIdx.y;
  const int* rowList = listBase ? (listBase + (size_t)e * Tn) : nullptr;
  const int count = countBase ? countBase[e] : Tn;
  const float* Wd = Wd0 + (size_t)e * expertStride;

  const int Ntiles = Hdim / 128;                    // 16
  int mt = (int)blockIdx.x / Ntiles;
  int nt = (int)blockIdx.x % Ntiles;
  int m0 = mt * 128;
  if (m0 >= count) return;
  int n0 = nt * 128;

  int tid = threadIdx.x;
  if (tid < 128) {
    int gr = m0 + tid;
    rIdx[tid] = (gr < count) ? (rowList ? rowList[gr] : gr)
                             : (rowList ? rowList[0] : 0);
  }
  __syncthreads();

  auto load_tiles = [&](int kt, int buf) {
    const int k0 = kt * 32;
    ushort_t* A = lA + buf * (128 * 32);
    ushort_t* B = lB + buf * (128 * 32);
#pragma unroll
    for (int it = 0; it < 2; ++it) {
      int idx = tid + it * 256;
      int m = idx >> 2, q = (idx & 3) * 8;
      *(uint4*)(A + m * 32 + q) =
          *(const uint4*)(Hm + (size_t)rIdx[m] * Idim + (size_t)(k0 + q));
    }
#pragma unroll
    for (int it = 0; it < 4; ++it) {
      int idx = tid + it * 256;
      int k = idx >> 5, n4 = (idx & 31) * 4;
      f4 v = __builtin_nontemporal_load(
          (const f4*)(Wd + (size_t)(k0 + k) * Hdim + (size_t)(n0 + n4)));
      ushort_t* b = B + n4 * 32 + k;
      b[0] = f2bf(v.x); b[32] = f2bf(v.y); b[64] = f2bf(v.z); b[96] = f2bf(v.w);
    }
  };

  int wave = tid >> 5, lane = tid & 31;
  int wm = wave & 3, wn = wave >> 2;
  int g = lane >> 4, ln = lane & 15;

  const v8f vzero = {};
  v8f C[2][4];
#pragma unroll
  for (int i = 0; i < 2; ++i)
#pragma unroll
    for (int j = 0; j < 4; ++j) C[i][j] = vzero;

  const int KT = Idim / 32;                         // 128 K-steps
  load_tiles(0, 0);
  __syncthreads();
  for (int kt = 0; kt < KT; ++kt) {
    int cur = kt & 1;
    if (kt + 1 < KT) load_tiles(kt + 1, cur ^ 1);

    const ushort_t* A = lA + cur * (128 * 32);
    const ushort_t* B = lB + cur * (128 * 32);

    Frag16 b[4];
#pragma unroll
    for (int j = 0; j < 4; ++j) {
      int nc = wn * 64 + j * 16 + ln;
      const uint4* pb = (const uint4*)(B + nc * 32);
      b[j].q[0] = pb[2 * g]; b[j].q[1] = pb[2 * g + 1];
    }
#pragma unroll
    for (int i2 = 0; i2 < 2; ++i2) {
      Frag16 a;
      int mr = wm * 32 + i2 * 16 + ln;
      const uint4* pa = (const uint4*)(A + mr * 32);
      a.q[0] = pa[2 * g]; a.q[1] = pa[2 * g + 1];
#pragma unroll
      for (int j = 0; j < 4; ++j) {
        C[i2][j] = __builtin_amdgcn_wmma_f32_16x16x32_bf16(
            false, a.v, false, b[j].v, (short)0, C[i2][j], false, false);
      }
    }
    __syncthreads();
  }

#pragma unroll
  for (int i2 = 0; i2 < 2; ++i2) {
    int mb = wm * 32 + i2 * 16 + g * 8;
#pragma unroll
    for (int r = 0; r < 8; ++r) {
      int m = mb + r;
      if (m0 + m < count) {
        int row = rIdx[m];
        size_t base = (size_t)row * Hdim + (size_t)(n0 + wn * 64 + ln);
#pragma unroll
        for (int j = 0; j < 4; ++j) {
          size_t o = base + (size_t)(j * 16);
          float v = C[i2][j][r];
          Out[o] = accumulate ? (Out[o] + v) : v;
        }
      }
    }
  }
}

// ---------------------------------------------------------------------------
// Host launcher. Workspace layout (bytes), total ~42 MB:
//   [0      ) eid    int[T]
//   [8192   ) sval   float[T]
//   [16384  ) counts int[8]
//   [16640  ) lists  int[E*T]
//   [82176  ) Xb     bf16[T*H]      (8 MB)   (16B aligned)
//   [+8MB   ) Hs     bf16[T*I]      (16 MB, shared-expert intermediate)
//   [+24MB  ) Hr     bf16[T*I]      (16 MB, routed intermediate, token-indexed)
// ---------------------------------------------------------------------------
extern "C" void kernel_launch(void* const* d_in, const int* in_sizes, int n_in,
                              void* d_out, int out_size, void* d_ws, size_t ws_size,
                              hipStream_t stream)
{
  (void)in_sizes; (void)n_in; (void)out_size; (void)ws_size;
  const float* hs  = (const float*)d_in[0];
  const float* rw  = (const float*)d_in[1];
  const float* gup = (const float*)d_in[2];
  const float* dwn = (const float*)d_in[3];
  const float* sg  = (const float*)d_in[4];
  const float* su  = (const float*)d_in[5];
  const float* sd  = (const float*)d_in[6];
  // d_in[7] = top_k (assumed 1, per reference setup)

  float* out    = (float*)d_out;
  float* scores = out + (size_t)Tn * Hdim;

  char* ws = (char*)d_ws;
  int*      eid    = (int*)(ws + 0);
  float*    sval   = (float*)(ws + 8192);
  int*      counts = (int*)(ws + 16384);
  int*      lists  = (int*)(ws + 16640);
  ushort_t* Xb     = (ushort_t*)(ws + 82176);
  ushort_t* Hs     = (ushort_t*)(ws + 82176 + (size_t)Tn * Hdim * 2);
  ushort_t* Hr     = (ushort_t*)(ws + 82176 + (size_t)Tn * Hdim * 2
                                          + (size_t)Tn * Idim * 2);

  router_kernel <<<Tn / 8, 256, 0, stream>>>(hs, rw, scores, eid, sval);
  compact_kernel<<<1,      256, 0, stream>>>(eid, lists, counts);
  cvt_kernel    <<<1024,   256, 0, stream>>>(hs, Xb, Tn * Hdim / 4);

  const int MT   = Tn / 128;       // 16 token tiles (worst-case per expert)
  const int NT_I = Idim / 128;     // 32
  const int NT_H = Hdim / 128;     // 16

  // shared expert: gate/up + SwiGLU -> Hs
  gateup_kernel<<<dim3(MT * NT_I, 1), 256, 0, stream>>>(
      Xb, sg, su, 0, Idim, Hs, nullptr, nullptr, nullptr);

  // routed experts (all 8 in one launch): gate/up + scaled SwiGLU -> Hr
  gateup_kernel<<<dim3(MT * NT_I, Ne), 256, 0, stream>>>(
      Xb, gup, gup + Idim, (size_t)Hdim * (size_t)(2 * Idim), 2 * Idim,
      Hr, lists, sval, counts);

  // shared down-proj writes out (covers all rows)
  down_kernel<<<dim3(MT * NT_H, 1), 256, 0, stream>>>(
      Hs, sd, 0, out, nullptr, nullptr, 0);

  // routed down-proj accumulates into out (all 8 experts in one launch)
  down_kernel<<<dim3(MT * NT_H, Ne), 256, 0, stream>>>(
      Hr, dwn, (size_t)Idim * (size_t)Hdim, out, lists, counts, 1);
}
